// ResidualBlock_78365973283133
// MI455X (gfx1250) — matrix operand order repaired and checked by reading, not compiled, not run on hardware
//
#include <hip/hip_runtime.h>
#include <cmath>

typedef __attribute__((ext_vector_type(16))) __bf16 v16bf;
typedef __attribute__((ext_vector_type(8)))  float  v8f;

#define XW 960       // row width of x / out
#define NW 8         // waves per block
#define ROWS_PER_BLOCK (NW * 16)

// ---------- helpers ----------

__device__ inline unsigned short f2bf(float f) {
  // round-to-nearest-even f32 -> bf16 bits
  unsigned u = __builtin_bit_cast(unsigned, f);
  u += 0x7FFFu + ((u >> 16) & 1u);
  return (unsigned short)(u >> 16);
}

union FragU {
  unsigned short s[16];
  uint4 q[2];
  v16bf v;
};

__device__ inline v8f wmma_bf16(v16bf a, v16bf b, v8f c) {
  // D = A(16x32) * B(32x16) + C, f32 accumulate
  return __builtin_amdgcn_wmma_f32_16x16x32_bf16(
      /*neg_a=*/false, a, /*neg_b=*/false, b,
      /*c_mod=*/(short)0, c, /*reuse_a=*/false, /*reuse_b=*/false);
}

// A fragment (16x32 bf16). Element (r, kk) = p[r*rstride + kk*kstride] (f32 source).
// Layout: lane l: r=l&15, hi=l>>4; elems 0..7 -> K=8*hi+e, elems 8..15 -> K=16+8*hi+(e-8)
__device__ inline v16bf load_a_f32(const float* __restrict__ p, int rstride, int kstride, int lane) {
  const int r = lane & 15, hi = lane >> 4;
  const float* row = p + r * rstride + 8 * hi * kstride;
  const float* row2 = row + 16 * kstride;
  FragU u;
#pragma unroll
  for (int e = 0; e < 8; ++e) u.s[e] = f2bf(row[e * kstride]);
#pragma unroll
  for (int e = 0; e < 8; ++e) u.s[8 + e] = f2bf(row2[e * kstride]);
  return u.v;
}

// A fragment from LDS bf16 row-major [16][S] starting at column k0
__device__ inline v16bf load_a_lds(const unsigned short* buf, int S, int k0, int lane) {
  const int r = lane & 15, hi = lane >> 4;
  const unsigned short* q = buf + r * S + k0 + 8 * hi;
  FragU u;
  u.q[0] = *(const uint4*)q;         // K = k0+8hi .. +7
  u.q[1] = *(const uint4*)(q + 16);  // K = k0+16+8hi .. +7
  return u.v;
}

// B fragment (32x16) from transposed bf16 weights Wt[Ntot][Ktot]
// lane l: n = n0 + (l&15), K = k0+8*(l>>4) .. +7, then k0+16+8*(l>>4) .. +7
__device__ inline v16bf load_b(const unsigned short* __restrict__ Wt, int Ktot, int n0, int k0, int lane) {
  const int n = n0 + (lane & 15);
  const int kb = k0 + 8 * (lane >> 4);
  const unsigned short* q = Wt + n * Ktot + kb;
  FragU u;
  u.q[0] = *(const uint4*)q;
  u.q[1] = *(const uint4*)(q + 16);
  return u.v;
}

__device__ inline float ssp(float v, float c) {
  // C_SSP * (softplus(v) - log 2), numerically stable softplus
  float sp = fmaxf(v, 0.0f) + log1pf(__expf(-fabsf(v)));
  return c * (sp - 0.69314718055994530942f);
}

// Store one 16x16 D tile with residual: out[m, colbase + (n0+n)*cstride] = x + acc*scale
__device__ inline void store_tile(float* __restrict__ out, const float* __restrict__ x,
                                  int row0, v8f acc, float scale,
                                  int colbase, int n0, int cstride, int lane) {
  const int n = lane & 15, hi = lane >> 4;
  const int col = colbase + (n0 + n) * cstride;
#pragma unroll
  for (int v = 0; v < 8; ++v) {
    const int m = row0 + 8 * hi + v;
    out[(size_t)m * XW + col] = x[(size_t)m * XW + col] + acc[v] * scale;
  }
}

// ---------- prologue: f32 [K][N] -> bf16 [N][K] ----------

__global__ void transpose_bf16_kernel(const float* __restrict__ src,
                                      unsigned short* __restrict__ dst, int K, int N) {
  int t = blockIdx.x * blockDim.x + threadIdx.x;
  if (t >= K * N) return;
  int nn = t / K, kk = t - nn * K;
  dst[t] = f2bf(src[(size_t)kk * N + nn]);
}

// ---------- main fused kernel: one wave = 16 rows ----------

__global__ __launch_bounds__(256) void resblock_kernel(
    const float* __restrict__ x, float* __restrict__ out,
    const unsigned short* __restrict__ W1s, const unsigned short* __restrict__ W1v1,
    const unsigned short* __restrict__ W1v2, const unsigned short* __restrict__ W2s,
    const unsigned short* __restrict__ W2v1, const unsigned short* __restrict__ W2v2,
    float cssp) {
  constexpr int SS = 256 + 8;  // padded LDS row strides (bank-conflict avoidance)
  constexpr int SV = 128 + 8;
  __shared__ __attribute__((aligned(16))) unsigned short ldsS[NW][16][SS];
  __shared__ __attribute__((aligned(16))) unsigned short ldsV[NW][16][SV];

  const int lane = threadIdx.x & 31;
  const int wave = threadIdx.x >> 5;
  const int row0 = (blockIdx.x * NW + wave) * 16;
  unsigned short* sbuf = &ldsS[wave][0][0];
  unsigned short* vbuf = &ldsV[wave][0][0];
  const float* xrow = x + (size_t)row0 * XW;
  const int n = lane & 15, hi = lane >> 4;

  const float SC0 = 0.0625f;               // 1/sqrt(256)
  const float SC1 = 0.088388347648318447f; // 1/sqrt(128)
  const float SC2 = 0.125f;                // 1/sqrt(64)

  // ---- phase 1: s = x0 @ W1s / 16 ; scalars -> LDS(bf16), gates -> regs ----
  v16bf a0[8];
#pragma unroll
  for (int ks = 0; ks < 8; ++ks) a0[ks] = load_a_f32(xrow + 32 * ks, XW, 1, lane);

  for (int t = 0; t < 16; ++t) {  // scalar tiles
    v8f acc = {};
#pragma unroll
    for (int ks = 0; ks < 8; ++ks)
      acc = wmma_bf16(a0[ks], load_b(W1s, 256, t * 16, 32 * ks, lane), acc);
#pragma unroll
    for (int v = 0; v < 8; ++v)
      sbuf[(8 * hi + v) * SS + t * 16 + n] = f2bf(ssp(acc[v] * SC0, cssp));
  }

  v8f gates[12];  // gate tiles stay in registers (same layout as v1/v2 D frags)
#pragma unroll
  for (int t = 0; t < 12; ++t) {
    v8f acc = {};
#pragma unroll
    for (int ks = 0; ks < 8; ++ks)
      acc = wmma_bf16(a0[ks], load_b(W1s, 256, (16 + t) * 16, 32 * ks, lane), acc);
#pragma unroll
    for (int v = 0; v < 8; ++v) gates[t][v] = ssp(acc[v] * SC0, cssp);
  }
  asm volatile("s_wait_dscnt 0" ::: "memory");

  // ---- y0 = scalars @ W2s / 16 (+ residual) ----
  v16bf as[8];
#pragma unroll
  for (int ks = 0; ks < 8; ++ks) as[ks] = load_a_lds(sbuf, SS, 32 * ks, lane);
  for (int t = 0; t < 16; ++t) {
    v8f acc = {};
#pragma unroll
    for (int ks = 0; ks < 8; ++ks)
      acc = wmma_bf16(as[ks], load_b(W2s, 256, t * 16, 32 * ks, lane), acc);
    store_tile(out, x, row0, acc, SC0, 0, t * 16, 1, lane);
  }

  // ---- v1 / y1 per channel (3 channels, K=128, x stride 3) ----
  for (int c = 0; c < 3; ++c) {
    v16bf a1[4];
#pragma unroll
    for (int ks = 0; ks < 4; ++ks)
      a1[ks] = load_a_f32(xrow + 256 + c + 96 * ks, XW, 3, lane);
#pragma unroll
    for (int t = 0; t < 8; ++t) {
      v8f acc = {};
#pragma unroll
      for (int ks = 0; ks < 4; ++ks)
        acc = wmma_bf16(a1[ks], load_b(W1v1, 128, t * 16, 32 * ks, lane), acc);
#pragma unroll
      for (int v = 0; v < 8; ++v)
        vbuf[(8 * hi + v) * SV + t * 16 + n] = f2bf(acc[v] * SC1 * gates[t][v]);
    }
    asm volatile("s_wait_dscnt 0" ::: "memory");
    v16bf av[4];
#pragma unroll
    for (int ks = 0; ks < 4; ++ks) av[ks] = load_a_lds(vbuf, SV, 32 * ks, lane);
    for (int t = 0; t < 8; ++t) {
      v8f acc = {};
#pragma unroll
      for (int ks = 0; ks < 4; ++ks)
        acc = wmma_bf16(av[ks], load_b(W2v1, 128, t * 16, 32 * ks, lane), acc);
      store_tile(out, x, row0, acc, SC1, 256 + c, t * 16, 3, lane);
    }
    asm volatile("s_wait_dscnt 0" ::: "memory");
  }

  // ---- v2 / y2 per channel (5 channels, K=64, x stride 5) ----
  for (int c = 0; c < 5; ++c) {
    v16bf a2[2];
#pragma unroll
    for (int ks = 0; ks < 2; ++ks)
      a2[ks] = load_a_f32(xrow + 640 + c + 160 * ks, XW, 5, lane);
#pragma unroll
    for (int t = 0; t < 4; ++t) {
      v8f acc = {};
#pragma unroll
      for (int ks = 0; ks < 2; ++ks)
        acc = wmma_bf16(a2[ks], load_b(W1v2, 64, t * 16, 32 * ks, lane), acc);
#pragma unroll
      for (int v = 0; v < 8; ++v)
        vbuf[(8 * hi + v) * SV + t * 16 + n] = f2bf(acc[v] * SC2 * gates[8 + t][v]);
    }
    asm volatile("s_wait_dscnt 0" ::: "memory");
    v16bf av[2];
#pragma unroll
    for (int ks = 0; ks < 2; ++ks) av[ks] = load_a_lds(vbuf, SV, 32 * ks, lane);
    for (int t = 0; t < 4; ++t) {
      v8f acc = {};
#pragma unroll
      for (int ks = 0; ks < 2; ++ks)
        acc = wmma_bf16(av[ks], load_b(W2v2, 64, t * 16, 32 * ks, lane), acc);
      store_tile(out, x, row0, acc, SC2, 640 + c, t * 16, 5, lane);
    }
    asm volatile("s_wait_dscnt 0" ::: "memory");
  }
}

// ---------- launch ----------

extern "C" void kernel_launch(void* const* d_in, const int* in_sizes, int n_in,
                              void* d_out, int out_size, void* d_ws, size_t ws_size,
                              hipStream_t stream) {
  (void)n_in; (void)out_size; (void)ws_size;
  const float* x    = (const float*)d_in[0];
  const float* w1s  = (const float*)d_in[1];
  const float* w1v1 = (const float*)d_in[2];
  const float* w1v2 = (const float*)d_in[3];
  const float* w2s  = (const float*)d_in[4];
  const float* w2v1 = (const float*)d_in[5];
  const float* w2v2 = (const float*)d_in[6];
  float* out = (float*)d_out;

  // bf16 transposed weights in workspace (256B-aligned offsets, total 442368 B)
  char* ws = (char*)d_ws;
  unsigned short* W1s  = (unsigned short*)(ws + 0);       // 448*256
  unsigned short* W1v1 = (unsigned short*)(ws + 229376);  // 128*128
  unsigned short* W1v2 = (unsigned short*)(ws + 262144);  // 64*64
  unsigned short* W2s  = (unsigned short*)(ws + 270336);  // 256*256
  unsigned short* W2v1 = (unsigned short*)(ws + 401408);  // 128*128
  unsigned short* W2v2 = (unsigned short*)(ws + 434176);  // 64*64

  auto tp = [&](const float* s, unsigned short* d, int K, int N) {
    int tot = K * N;
    transpose_bf16_kernel<<<(tot + 255) / 256, 256, 0, stream>>>(s, d, K, N);
  };
  tp(w1s,  W1s,  256, 448);
  tp(w1v1, W1v1, 128, 128);
  tp(w1v2, W1v2, 64, 64);
  tp(w2s,  W2s,  256, 256);
  tp(w2v1, W2v1, 128, 128);
  tp(w2v2, W2v2, 64, 64);

  // C_SSP = 1/sqrt(trapz(f^2 * phi)) over 200001 points in [-10,10] (matches reference)
  double sum = 0.0, prev = 0.0;
  const int NPT = 200001;
  const double h = 20.0 / (NPT - 1);
  for (int i = 0; i < NPT; ++i) {
    double z = -10.0 + h * i;
    double f = log1p(exp(z)) - 0.69314718055994530942;
    double phi = exp(-0.5 * z * z) * 0.39894228040143267794;
    double val = f * f * phi;
    if (i) sum += 0.5 * (prev + val) * h;
    prev = val;
  }
  float cssp = (float)(1.0 / sqrt(sum));

  int nrows = in_sizes[0] / XW;                 // 65536
  int nblocks = nrows / ROWS_PER_BLOCK;         // 512
  resblock_kernel<<<nblocks, 256, 0, stream>>>(x, out, W1s, W1v1, W1v2,
                                               W2s, W2v1, W2v2, cssp);
}
